// SelfAttentionHead_38843684225220
// MI455X (gfx1250) — matrix operand order repaired and checked by reading, not compiled, not run on hardware
//
#include <hip/hip_runtime.h>

// ---------------------------------------------------------------------------
// Causal self-attention head, CDNA5 (gfx1250, wave32).
// bf16 WMMA (fp32 accum) + double-buffered GLOBAL_LOAD_ASYNC_TO_LDS staging.
// B=8, T=2048, D=1024, H=128.
// ---------------------------------------------------------------------------

#define B_ 8
#define T_ 2048
#define D_ 1024
#define H_ 128
#define BT_ (B_ * T_)

#define KSTR 136   // ushorts per K LDS row (272 B = 17*16, conflict-padded)
#define VSTR 72    // ushorts per V LDS row (144 B =  9*16, conflict-padded)

typedef __attribute__((ext_vector_type(16))) __bf16 v16bf;
typedef __attribute__((ext_vector_type(8)))  float  v8f;

union FragB {            // 16 bf16 = 8 dwords
    v16bf v;
    unsigned u[8];
};

__device__ __forceinline__ unsigned short f2bf(float f) {
    unsigned u = __float_as_uint(f);
    unsigned r = u + 0x7FFFu + ((u >> 16) & 1u);   // round-to-nearest-even
    return (unsigned short)(r >> 16);
}

// ===========================================================================
// Kernel 1: projections. Q,K stored [b][t][h] bf16; V stored TRANSPOSED
// [b][h][t] bf16 so the attention kernel can async-copy V tiles directly.
// grid = (BT/64, 3), block = 128 (4 waves).
// ===========================================================================
__global__ __launch_bounds__(128) void proj_kernel(
    const float* __restrict__ x,
    const float* __restrict__ Wq, const float* __restrict__ Wk,
    const float* __restrict__ Wv,
    unsigned short* __restrict__ Qb, unsigned short* __restrict__ Kb,
    unsigned short* __restrict__ Vb)
{
    __shared__ unsigned short ldsX[64 * 68];    // [row][k], pad 64->68
    __shared__ unsigned short ldsW[128 * 68];   // transposed [h][k], pad

    const float* W = (blockIdx.y == 0) ? Wq : (blockIdx.y == 1) ? Wk : Wv;
    unsigned short* O = (blockIdx.y == 0) ? Qb : (blockIdx.y == 1) ? Kb : Vb;

    const int r0   = blockIdx.x * 64;
    const int tid  = threadIdx.x;
    const int lane = tid & 31;
    const int wv   = tid >> 5;
    const int l16  = lane & 15;
    const int half = lane >> 4;

    v8f acc[8];
    #pragma unroll
    for (int nt = 0; nt < 8; nt++) acc[nt] = v8f{0,0,0,0,0,0,0,0};

    for (int kc = 0; kc < D_; kc += 64) {
        __syncthreads();
        #pragma unroll 4
        for (int i = 0; i < 32; i++) {                // x tile 64x64
            int idx = tid + i * 128;
            int rr = idx >> 6, cc = idx & 63;
            ldsX[rr * 68 + cc] = f2bf(x[(size_t)(r0 + rr) * D_ + kc + cc]);
        }
        #pragma unroll 4
        for (int kk = 0; kk < 64; kk++) {             // W tile 64x128 (T)
            ldsW[tid * 68 + kk] = f2bf(W[(size_t)(kc + kk) * H_ + tid]);
        }
        __syncthreads();

        FragB a[2];
        {
            const unsigned* px = (const unsigned*)&ldsX[(16 * wv + l16) * 68];
            #pragma unroll
            for (int ks = 0; ks < 2; ks++)
                #pragma unroll
                for (int v = 0; v < 8; v++) {
                    int kA = 8 * half + 2 * v + ((v >= 4) ? 8 : 0);
                    a[ks].u[v] = px[(32 * ks + kA) >> 1];
                }
        }
        #pragma unroll
        for (int nt = 0; nt < 8; nt++) {
            const unsigned* pw = (const unsigned*)&ldsW[(16 * nt + l16) * 68];
            #pragma unroll
            for (int ks = 0; ks < 2; ks++) {
                FragB bfr;
                #pragma unroll
                for (int v = 0; v < 8; v++) {
                    int kB = 8 * half + 2 * v + ((v >= 4) ? 8 : 0);
                    bfr.u[v] = pw[(32 * ks + kB) >> 1];
                }
                acc[nt] = __builtin_amdgcn_wmma_f32_16x16x32_bf16(
                    false, a[ks].v, false, bfr.v, (short)0, acc[nt],
                    false, false);
            }
        }
    }

    if (blockIdx.y == 2) {                 // V: transposed store [b][h][t]
        #pragma unroll
        for (int nt = 0; nt < 8; nt++)
            #pragma unroll
            for (int v = 0; v < 8; v++) {
                int row = r0 + 16 * wv + v + 8 * half;   // global t index
                int col = 16 * nt + l16;                 // h index
                int bb  = row >> 11;                     // / T_
                int tt  = row & (T_ - 1);
                O[((size_t)bb * H_ + col) * T_ + tt] = f2bf(acc[nt][v]);
            }
    } else {                               // Q,K: row-major [b][t][h]
        #pragma unroll
        for (int nt = 0; nt < 8; nt++)
            #pragma unroll
            for (int v = 0; v < 8; v++) {
                int row = r0 + 16 * wv + v + 8 * half;
                int col = 16 * nt + l16;
                O[(size_t)row * H_ + col] = f2bf(acc[nt][v]);
            }
    }
}

// ===========================================================================
// Kernel 2: causal flash attention. Double-buffered async K/V staging.
// grid = (T/64, B), block = 128 (4 waves); wave w owns Q rows q0+16w..+15.
// ===========================================================================
__global__ __launch_bounds__(128) void attn_kernel(
    const unsigned short* __restrict__ Qb,
    const unsigned short* __restrict__ Kb,
    const unsigned short* __restrict__ Vb,
    float* __restrict__ out)
{
    __shared__ __attribute__((aligned(16))) unsigned short ldsKb[2][64 * KSTR];
    __shared__ __attribute__((aligned(16))) unsigned short ldsVb[2][128 * VSTR];
    __shared__ unsigned short ldsP[4][16 * 66];   // per-wave P [row][key]

    const float SCALE = 0.08838834764831845f;     // 1/sqrt(H)
    const float L2E   = 1.4426950408889634f;

    const int b    = blockIdx.y;
    const int rb   = blockIdx.x;
    const int tid  = threadIdx.x;
    const int lane = tid & 31;
    const int wv   = tid >> 5;
    const int l16  = lane & 15;
    const int half = lane >> 4;

    const unsigned short* Qp = Qb + (size_t)b * T_ * H_;
    const unsigned short* Kp = Kb + (size_t)b * T_ * H_;
    const unsigned short* Vp = Vb + (size_t)b * H_ * T_;   // transposed [h][t]

    const int q0 = rb * 64;
    const int qw = q0 + wv * 16;

    // --- issue one KV block's async copies: 8 + 8 x B128 per thread ---
    auto issueKV = [&](int s0, int bufi) {
        const char* kg = (const char*)(Kp + (size_t)s0 * H_);   // 64 x 256B
        const char* vg = (const char*)(Vp + s0);                // 128 x 128B
        unsigned kbase = (unsigned)(uintptr_t)&ldsKb[bufi][0];
        unsigned vbase = (unsigned)(uintptr_t)&ldsVb[bufi][0];
        #pragma unroll
        for (int i = 0; i < 8; i++) {
            unsigned o   = (unsigned)((i * 128 + tid) * 16);    // 16 KB flat
            unsigned lds = kbase + (o >> 8) * (KSTR * 2) + (o & 255);
            asm volatile("global_load_async_to_lds_b128 %0, %1, %2"
                         :: "v"(lds), "v"(o), "s"(kg) : "memory");
        }
        #pragma unroll
        for (int i = 0; i < 8; i++) {
            unsigned o   = (unsigned)((i * 128 + tid) * 16);
            unsigned go  = (o >> 7) * (T_ * 2) + (o & 127);     // h*rowstride
            unsigned lds = vbase + (o >> 7) * (VSTR * 2) + (o & 127);
            asm volatile("global_load_async_to_lds_b128 %0, %1, %2"
                         :: "v"(lds), "v"(go), "s"(vg) : "memory");
        }
    };

    // Q A-fragments: registers, reused across whole KV loop
    FragB qf[4];
    {
        const unsigned* pq = (const unsigned*)(Qp + (size_t)(qw + l16) * H_);
        #pragma unroll
        for (int f = 0; f < 4; f++)
            #pragma unroll
            for (int v = 0; v < 8; v++) {
                int kA = 8 * half + 2 * v + ((v >= 4) ? 8 : 0);
                qf[f].u[v] = pq[(32 * f + kA) >> 1];
            }
    }

    float m[8], l[8];
    v8f acc[8];
    #pragma unroll
    for (int v = 0; v < 8; v++) { m[v] = -3.0e38f; l[v] = 0.0f; }
    #pragma unroll
    for (int nt = 0; nt < 8; nt++) acc[nt] = v8f{0,0,0,0,0,0,0,0};

    const int nkb = rb + 1;                  // causal: key blocks 0..rb
    issueKV(0, 0);                           // prologue: stage block 0

    for (int kb = 0; kb < nkb; kb++) {
        const int s0  = kb * 64;
        const int cur = kb & 1;

        if (kb + 1 < nkb) {                  // stage next block, then wait
            issueKV(s0 + 64, cur ^ 1);       // for the OLDER 16 asyncs
            asm volatile("s_wait_asynccnt 0x10" ::: "memory");
        } else {
            asm volatile("s_wait_asynccnt 0x0" ::: "memory");
        }
        __syncthreads();                     // all waves' data resident

        const unsigned short* ldsK = &ldsKb[cur][0];
        const unsigned short* ldsV = &ldsVb[cur][0];

        // --- S = Q @ K^T : 4 column tiles x 4 K-steps ---
        v8f sacc[4];
        #pragma unroll
        for (int t = 0; t < 4; t++) sacc[t] = v8f{0,0,0,0,0,0,0,0};
        #pragma unroll
        for (int t = 0; t < 4; t++) {
            const unsigned* pk =
                (const unsigned*)&ldsK[(16 * t + l16) * KSTR];
            #pragma unroll
            for (int ks = 0; ks < 4; ks++) {
                FragB kf;
                #pragma unroll
                for (int v = 0; v < 8; v++) {
                    int kB = 8 * half + 2 * v + ((v >= 4) ? 8 : 0);
                    kf.u[v] = pk[(32 * ks + kB) >> 1];
                }
                sacc[t] = __builtin_amdgcn_wmma_f32_16x16x32_bf16(
                    false, qf[ks].v, false, kf.v, (short)0, sacc[t],
                    false, false);
            }
        }

        // --- causal mask + scale, row max ---
        float nm[8];
        #pragma unroll
        for (int v = 0; v < 8; v++) nm[v] = m[v];
        #pragma unroll
        for (int t = 0; t < 4; t++) {
            int kj = s0 + 16 * t + l16;
            #pragma unroll
            for (int v = 0; v < 8; v++) {
                int qi = qw + v + 8 * half;
                float s = sacc[t][v] * SCALE;
                s = (kj <= qi) ? s : -3.0e38f;
                sacc[t][v] = s;
                nm[v] = fmaxf(nm[v], s);
            }
        }
        #pragma unroll
        for (int v = 0; v < 8; v++) {
            float xv = nm[v];
            xv = fmaxf(xv, __shfl_xor(xv, 1, 16));
            xv = fmaxf(xv, __shfl_xor(xv, 2, 16));
            xv = fmaxf(xv, __shfl_xor(xv, 4, 16));
            xv = fmaxf(xv, __shfl_xor(xv, 8, 16));
            nm[v] = xv;
        }
        float alpha[8], rs[8];
        #pragma unroll
        for (int v = 0; v < 8; v++) {
            alpha[v] = exp2f((m[v] - nm[v]) * L2E);
            m[v] = nm[v];
            rs[v] = 0.0f;
        }

        // --- P = exp(S - m) -> bf16 in per-wave LDS (A-layout source) ---
        unsigned short* pP = &ldsP[wv][0];
        #pragma unroll
        for (int t = 0; t < 4; t++) {
            #pragma unroll
            for (int v = 0; v < 8; v++) {
                float p = exp2f((sacc[t][v] - m[v]) * L2E);
                rs[v] += p;
                pP[(v + 8 * half) * 66 + 16 * t + l16] = f2bf(p);
            }
        }
        #pragma unroll
        for (int v = 0; v < 8; v++) {
            float xv = rs[v];
            xv += __shfl_xor(xv, 1, 16);
            xv += __shfl_xor(xv, 2, 16);
            xv += __shfl_xor(xv, 4, 16);
            xv += __shfl_xor(xv, 8, 16);
            l[v] = l[v] * alpha[v] + xv;
        }
        #pragma unroll
        for (int nt = 0; nt < 8; nt++)
            #pragma unroll
            for (int v = 0; v < 8; v++)
                acc[nt][v] *= alpha[v];

        // same-wave LDS RAW on P before fragment loads
        asm volatile("s_wait_dscnt 0" ::: "memory");

        // --- acc += P @ V : 2 K-steps x 8 N-tiles ---
        #pragma unroll
        for (int ks = 0; ks < 2; ks++) {
            FragB af;
            const unsigned* pa = (const unsigned*)&pP[l16 * 66];
            #pragma unroll
            for (int v = 0; v < 8; v++) {
                int kA = 8 * half + 2 * v + ((v >= 4) ? 8 : 0);
                af.u[v] = pa[(32 * ks + kA) >> 1];
            }
            #pragma unroll
            for (int nt = 0; nt < 8; nt++) {
                const unsigned* pv =
                    (const unsigned*)&ldsV[(16 * nt + l16) * VSTR];
                FragB vf;
                #pragma unroll
                for (int v = 0; v < 8; v++) {
                    int kB = 8 * half + 2 * v + ((v >= 4) ? 8 : 0);
                    vf.u[v] = pv[(32 * ks + kB) >> 1];
                }
                acc[nt] = __builtin_amdgcn_wmma_f32_16x16x32_bf16(
                    false, af.v, false, vf.v, (short)0, acc[nt],
                    false, false);
            }
        }

        // all our LDS reads retired before next iteration's asyncs overwrite
        asm volatile("s_wait_dscnt 0" ::: "memory");
        __syncthreads();
    }

    // --- epilogue: out = acc / l ---
    float* op = out + ((size_t)b * T_ + qw) * H_;
    float inv[8];
    #pragma unroll
    for (int v = 0; v < 8; v++) inv[v] = 1.0f / l[v];
    #pragma unroll
    for (int nt = 0; nt < 8; nt++)
        #pragma unroll
        for (int v = 0; v < 8; v++)
            op[(size_t)(v + 8 * half) * H_ + 16 * nt + l16] =
                acc[nt][v] * inv[v];
}

// ===========================================================================
extern "C" void kernel_launch(void* const* d_in, const int* in_sizes, int n_in,
                              void* d_out, int out_size, void* d_ws,
                              size_t ws_size, hipStream_t stream)
{
    (void)in_sizes; (void)n_in; (void)out_size; (void)ws_size;
    const float* x  = (const float*)d_in[0];
    const float* Wq = (const float*)d_in[1];
    const float* Wk = (const float*)d_in[2];
    const float* Wv = (const float*)d_in[3];

    unsigned short* Qb = (unsigned short*)d_ws;           // BT*H bf16
    unsigned short* Kb = Qb + (size_t)BT_ * H_;
    unsigned short* Vb = Kb + (size_t)BT_ * H_;           // stored [b][h][t]

    proj_kernel<<<dim3(BT_ / 64, 3), 128, 0, stream>>>(x, Wq, Wk, Wv,
                                                       Qb, Kb, Vb);
    attn_kernel<<<dim3(T_ / 64, B_), 128, 0, stream>>>(Qb, Kb, Vb,
                                                       (float*)d_out);
}
